// RecurrentEmbeddingModel_28432683499536
// MI455X (gfx1250) — matrix operand order repaired and checked by reading, not compiled, not run on hardware
//
#include <hip/hip_runtime.h>
#include <hip/hip_bf16.h>

typedef __attribute__((ext_vector_type(16))) __bf16    v16bf;
typedef __attribute__((ext_vector_type(8)))  float     v8f;
typedef __attribute__((ext_vector_type(2)))  float     v2f;
typedef __attribute__((ext_vector_type(8)))  unsigned  v8u;
typedef __attribute__((ext_vector_type(4)))  unsigned  u32x4;
typedef __attribute__((ext_vector_type(8)))  int       i32x8;
typedef __attribute__((ext_vector_type(4)))  int       i32x4;

#define NN   8192
#define RR   16
#define LL   256
#define HID  32
#define XSTRIDE 260                 // padded row stride (floats): bank-conflict-free, 16B aligned
#define WAVE_LDS (16 * XSTRIDE)     // 4160 floats per wave
#define WPB  2                      // waves per block -> 33280 B LDS per block

#if defined(__has_builtin)
# if __has_builtin(__builtin_amdgcn_wmma_f32_16x16x4_f32)
#  define USE_WMMA_X4 1
# endif
# if __has_builtin(__builtin_amdgcn_perm)
#  define USE_PERM_PK 1
# endif
# if __has_builtin(__builtin_amdgcn_tensor_load_to_lds) && __has_builtin(__builtin_amdgcn_s_wait_tensorcnt)
#  define USE_TDM 1
# endif
#endif
#ifndef USE_WMMA_X4
# define USE_WMMA_X4 0
#endif
#ifndef USE_PERM_PK
# define USE_PERM_PK 0
#endif
#ifndef USE_TDM
# define USE_TDM 0
#endif

// Branch-free tanh fallback: tanh(x) = 1 - 2/(e^(2x)+1) via v_exp_f32/v_rcp_f32.
__device__ __forceinline__ float tanh_fallback(float x) {
    float e = __builtin_amdgcn_exp2f(x * 2.885390081777927f);   // e^(2x)
    float r = __builtin_amdgcn_rcpf(e + 1.0f);
    return fmaf(-2.0f, r, 1.0f);
}
// Hardware v_tanh_f32 TRANS op (confirmed emitted on this toolchain).
#if defined(__has_builtin)
# if __has_builtin(__builtin_amdgcn_tanhf)
#  define TANHF(x) __builtin_amdgcn_tanhf(x)
# elif __has_builtin(__builtin_amdgcn_tanh_f32)
#  define TANHF(x) __builtin_amdgcn_tanh_f32(x)
# endif
#endif
#ifndef TANHF
# define TANHF(x) tanh_fallback(x)
#endif

// Pack two f32 -> {bf16(hi)[31:16], bf16(lo)[15:0]} in ONE v_perm_b32.
__device__ __forceinline__ unsigned pk_bf16(float lo, float hi) {
#if USE_PERM_PK
    return __builtin_amdgcn_perm(__float_as_uint(hi), __float_as_uint(lo), 0x07060302u);
#else
    unsigned ul = __float_as_uint(lo), uh = __float_as_uint(hi);
    return (uh & 0xFFFF0000u) | (ul >> 16);
#endif
}

__global__ __launch_bounds__(WPB * 32)
void rnn_embed_kernel(const float* __restrict__ h,  const int* __restrict__ l,
                      const float* __restrict__ w_ih, const float* __restrict__ w_hh,
                      const float* __restrict__ b_ih, const float* __restrict__ b_hh,
                      const float* __restrict__ w1, const float* __restrict__ b1,
                      const float* __restrict__ w2, const float* __restrict__ b2,
                      const float* __restrict__ w3, const float* __restrict__ b3,
                      float* __restrict__ out)
{
    __shared__ float smem[WPB * WAVE_LDS];
    const int lane = threadIdx.x & 31;
    const int wave = threadIdx.x >> 5;
    const int tile = blockIdx.x * WPB + wave;      // == sample index n
    float* xws = smem + wave * WAVE_LDS;

    const int  m16  = lane & 15;
    const int  half = lane >> 4;
    const bool lo   = (half == 0);

    // ---- stage this sample's 16 sequence rows (16 x 256 f32) into padded LDS ----
#if USE_TDM
    {
        // Tensor Data Mover: 2D tile 256x16 f32, dim0 stride 256 elements,
        // LDS pad 4 DWORDs every 256 DWORDs == XSTRIDE 260 floats per row.
        unsigned lds_base = (unsigned)(size_t)(void*)xws;   // low bits = LDS offset
        unsigned long long ga =
            (unsigned long long)(size_t)(const void*)(h + (size_t)tile * (RR * LL));
        u32x4 g0 = { 1u,                                    // count=1 (user descriptor)
                     lds_base,                              // lds_addr [63:32]
                     (unsigned)(ga & 0xFFFFFFFFu),          // global_addr[31:0]
                     (unsigned)((ga >> 32) & 0x01FFFFFFu) | (2u << 30) }; // addr[56:32]|type=2
        i32x8 g1 = { (2 << 16) | (1 << 20) | (7 << 22) | (3 << 25),
                     //  data_size=4B | pad_enable | pad_interval=256dw | pad_amount=4dw
                     (int)(256u << 16),                     // tensor_dim0 = 256  @[79:48]
                     (int)(16u  << 16),                     // tensor_dim1 = 16   @[111:80]
                     (int)(256u << 16),                     // tile_dim0  = 256   @[127:112]
                     16,                                    // tile_dim1  = 16    @[143:128]
                     256,                                   // tensor_dim0_stride @[207:160]
                     0, 0 };
        i32x4 gz = { 0, 0, 0, 0 };
#if __clang_major__ >= 23
        i32x8 gz8 = { 0, 0, 0, 0, 0, 0, 0, 0 };
        __builtin_amdgcn_tensor_load_to_lds(g0, g1, gz, gz, gz8, 0);
#else
        __builtin_amdgcn_tensor_load_to_lds(g0, g1, gz, gz, 0);
#endif
        __builtin_amdgcn_s_wait_tensorcnt(0);
    }
#else
    {
        const float4* src = (const float4*)(h + (size_t)tile * (RR * LL));
        #pragma unroll
        for (int it = 0; it < 32; ++it) {
            int c  = it * 32 + lane;     // 1024 float4 chunks total
            int r  = c >> 6;             // row 0..15
            int c4 = c & 63;             // float4 within row
            float4 v = src[r * 64 + c4];
            *(float4*)&xws[r * XSTRIDE + c4 * 4] = v;
        }
    }
#endif

    const int mylen = l[tile * RR + m16];

    // wave-uniform max length -> scalar loop trip count (lanes 16-31 mirror 0-15)
    int maxlen = mylen;
    maxlen = max(maxlen, __shfl_xor(maxlen, 1));
    maxlen = max(maxlen, __shfl_xor(maxlen, 2));
    maxlen = max(maxlen, __shfl_xor(maxlen, 4));
    maxlen = max(maxlen, __shfl_xor(maxlen, 8));
    maxlen = __builtin_amdgcn_readfirstlane(maxlen);

    // Feature-row permutation so D-layout -> next-step B-layout needs no shuffles:
    // top tile owns features {0..7,16..23}, bottom tile {8..15,24..31}.
    const int row_t = (m16 < 8) ? m16 : m16 + 8;
    const int row_b = row_t + 8;

    // ---- stationary bf16 A-matrices holding W_hh rows (WMMA 16x32 A layout) ----
    v8u atu, abu;
    #pragma unroll
    for (int j = 0; j < 8; ++j) {
        int kb = 2 * j + (half ? 16 : 0);
        atu[j] = pk_bf16(w_hh[row_t * HID + kb], w_hh[row_t * HID + kb + 1]);
        abu[j] = pk_bf16(w_hh[row_b * HID + kb], w_hh[row_b * HID + kb + 1]);
    }
    const v16bf a_t = __builtin_bit_cast(v16bf, atu);
    const v16bf a_b = __builtin_bit_cast(v16bf, abu);

#if USE_WMMA_X4
    // rank-1 + bias via V_WMMA_F32_16X16X4_F32: A col0 = w_ih, col1 = b_ih+b_hh
    v2f axt, axb;
    axt[0] = lo ? w_ih[row_t] : 0.0f;
    axt[1] = lo ? (b_ih[row_t] + b_hh[row_t]) : 0.0f;
    axb[0] = lo ? w_ih[row_b] : 0.0f;
    axb[1] = lo ? (b_ih[row_b] + b_hh[row_b]) : 0.0f;
    const float bx1 = lo ? 1.0f : 0.0f;
#else
    float wih_t[8], wih_b[8], bia_t[8], bia_b[8];
    #pragma unroll
    for (int v = 0; v < 8; ++v) {
        int ft = half ? (v + 16) : v;
        int fb = half ? (v + 24) : (v + 8);
        wih_t[v] = w_ih[ft];  bia_t[v] = b_ih[ft] + b_hh[ft];
        wih_b[v] = w_ih[fb];  bia_b[v] = b_ih[fb] + b_hh[fb];
    }
#endif

    v8f czero;
    #pragma unroll
    for (int i = 0; i < 8; ++i) czero[i] = 0.0f;

    v8u hu;                                    // hidden state, bf16-packed, B-layout
    #pragma unroll
    for (int i = 0; i < 8; ++i) hu[i] = 0u;

    // ================= recurrence: lockstep masked steps =================
    int j0 = mylen - 1; if (j0 < 0) j0 = 0;
    float xt = xws[m16 * XSTRIDE + j0];        // software-pipelined gather

    for (int t = 0; t < maxlen; ++t) {
        // issue next step's gather early; wait lands after the tanh block
        int jn = mylen - 2 - t; if (jn < 0) jn = 0;
        float xnext = xws[m16 * XSTRIDE + jn];

        v8f acc_t, acc_b;
#if USE_WMMA_X4
        v2f bx; bx[0] = lo ? xt : 0.0f; bx[1] = bx1;      // B row0 = x_t, row1 = 1
        acc_t = __builtin_amdgcn_wmma_f32_16x16x4_f32(false, axt, false, bx,
                                                      (short)0, czero, false, false);
        acc_b = __builtin_amdgcn_wmma_f32_16x16x4_f32(false, axb, false, bx,
                                                      (short)0, czero, false, false);
#else
        #pragma unroll
        for (int v = 0; v < 8; ++v) {
            acc_t[v] = fmaf(xt, wih_t[v], bia_t[v]);
            acc_b[v] = fmaf(xt, wih_b[v], bia_b[v]);
        }
#endif
        const v16bf hb = __builtin_bit_cast(v16bf, hu);
        acc_t = __builtin_amdgcn_wmma_f32_16x16x32_bf16(false, a_t, false, hb,
                                                        (short)0, acc_t, false, false);
        acc_b = __builtin_amdgcn_wmma_f32_16x16x32_bf16(false, a_b, false, hb,
                                                        (short)0, acc_b, false, false);

        const bool keep = (t < mylen);                    // per-lane mask, all vgprs
        #pragma unroll
        for (int v = 0; v < 4; ++v) {
            float t0 = TANHF(acc_t[2 * v]), t1 = TANHF(acc_t[2 * v + 1]);
            float u0 = TANHF(acc_b[2 * v]), u1 = TANHF(acc_b[2 * v + 1]);
            unsigned pt = pk_bf16(t0, t1);                // single v_perm_b32 each
            unsigned pb = pk_bf16(u0, u1);
            hu[v]     = keep ? pt : hu[v];
            hu[v + 4] = keep ? pb : hu[v + 4];
        }
        xt = xnext;
    }

    // ================= epilogue: emb -> 3-layer MLP (tiny) =================
    float* emb = xws;                                     // reuse staging space
    #pragma unroll
    for (int v = 0; v < 8; ++v) {
        int fb = ((v < 4) ? (2 * v) : (2 * (v - 4) + 8)) + half * 16;
        unsigned u = hu[v];
        float f0 = __uint_as_float(u << 16);
        float f1 = __uint_as_float(u & 0xFFFF0000u);
        *(float2*)&emb[m16 * HID + fb] = make_float2(f0, f1);
    }

    float* z1s = xws + 512;
    float* z2s = xws + 544;
    {   // layer 1: 32 outputs, 512-dim dot each; one output per lane
        float acc = b1[lane];
        const float4* w1v = (const float4*)w1 + lane * 128;
        const float4* ev  = (const float4*)emb;
        #pragma unroll 4
        for (int jj = 0; jj < 128; ++jj) {
            float4 w = w1v[jj]; float4 e = ev[jj];
            acc = fmaf(w.x, e.x, acc); acc = fmaf(w.y, e.y, acc);
            acc = fmaf(w.z, e.z, acc); acc = fmaf(w.w, e.w, acc);
        }
        z1s[lane] = fmaxf(acc, 0.0f);
    }
    {   // layer 2
        float acc = b2[lane];
        const float4* w2v = (const float4*)w2 + lane * 8;
        const float4* z1v = (const float4*)z1s;
        #pragma unroll
        for (int jj = 0; jj < 8; ++jj) {
            float4 w = w2v[jj]; float4 z = z1v[jj];
            acc = fmaf(w.x, z.x, acc); acc = fmaf(w.y, z.y, acc);
            acc = fmaf(w.z, z.z, acc); acc = fmaf(w.w, z.w, acc);
        }
        z2s[lane] = fmaxf(acc, 0.0f);
    }
    if (lane == 0) {
        float s = b3[0];
        #pragma unroll
        for (int jj = 0; jj < 32; ++jj) s = fmaf(w3[jj], z2s[jj], s);
        out[tile] = s;
    }
}

extern "C" void kernel_launch(void* const* d_in, const int* in_sizes, int n_in,
                              void* d_out, int out_size, void* d_ws, size_t ws_size,
                              hipStream_t stream) {
    (void)in_sizes; (void)n_in; (void)out_size; (void)d_ws; (void)ws_size;
    const float* h    = (const float*)d_in[0];
    const int*   l    = (const int*)  d_in[1];
    const float* w_ih = (const float*)d_in[2];
    const float* w_hh = (const float*)d_in[3];
    const float* b_ih = (const float*)d_in[4];
    const float* b_hh = (const float*)d_in[5];
    const float* w1   = (const float*)d_in[6];
    const float* b1   = (const float*)d_in[7];
    const float* w2   = (const float*)d_in[8];
    const float* b2   = (const float*)d_in[9];
    const float* w3   = (const float*)d_in[10];
    const float* b3   = (const float*)d_in[11];
    float* out = (float*)d_out;

    dim3 grid(NN / WPB);      // 4096 blocks, 2 waves (tiles) per block
    dim3 block(WPB * 32);
    rnn_embed_kernel<<<grid, block, 0, stream>>>(h, l, w_ih, w_hh, b_ih, b_hh,
                                                 w1, b1, w2, b2, w3, b3, out);
}
